// GPT_32684701122574
// MI455X (gfx1250) — matrix operand order repaired and checked by reading, not compiled, not run on hardware
//
#include <hip/hip_runtime.h>
#include <hip/hip_bf16.h>
#include <math.h>

typedef __attribute__((ext_vector_type(16))) _Float16 v16h;
typedef __attribute__((ext_vector_type(8)))  _Float16 v8h;
typedef __attribute__((ext_vector_type(8)))  float    v8f;
typedef __attribute__((ext_vector_type(4)))  unsigned int v4u;
typedef __attribute__((ext_vector_type(8)))  unsigned int v8u;

static constexpr int LAYERS = 12, HEADS = 12, DMODEL = 768, DFF = 3072, VOCAB = 32000;
static constexpr int HDIM = 64, SEQ = 1024, BATCH = 2, NTOK = BATCH * SEQ;   // 2048 tokens
static constexpr int QKVD = 3 * DMODEL;                                      // 2304

#define DEV __device__ __forceinline__

// ---- WMMA fragment helpers (layouts per CDNA5 ISA §7.12.2, wave32) ----------
DEV v16h make_a_frag(const _Float16* p) {
  v8h lo = *(const v8h*)p;
  v8h hi = *(const v8h*)(p + 8);
  v16h f;
#pragma unroll
  for (int i = 0; i < 8; ++i) { f[i] = lo[i]; f[i + 8] = hi[i]; }
  return f;
}

DEV v8f wmma_f16(v16h a, v16h b, v8f c) {
  return __builtin_amdgcn_wmma_f32_16x16x32_f16(false, a, false, b, (short)0, c,
                                                false, false);
}

// ---- CDNA5 async / TDM data movers (inline asm: portable across toolchains) -
DEV void async_copy_b128(unsigned int lds_off, const void* gaddr) {
  // GLOBAL_LOAD_ASYNC_TO_LDS_B128 (tracked by ASYNCcnt), GV addressing
  asm volatile("global_load_async_to_lds_b128 %0, %1, off"
               :: "v"(lds_off), "v"(gaddr) : "memory");
}
DEV void wait_asynccnt0() { asm volatile("s_wait_asynccnt 0" ::: "memory"); }

// Tensor Data Mover: 2D tile (tile_d1 rows x tile_d0 elems, f16) -> LDS.
// Descriptor per CDNA5 ISA (group0) / §8.4 (group1).
DEV void tdm_load_tile(unsigned int lds_off, const _Float16* g,
                       unsigned int tensor_d0, unsigned int tensor_d1,
                       unsigned int tile_d0, unsigned int tile_d1,
                       unsigned int stride0_elems) {
  const unsigned long long ga = (unsigned long long)(uintptr_t)g;
  v4u g0;
  g0[0] = 1u;                                                  // count=1, user D#
  g0[1] = lds_off;                                             // lds_addr (bytes)
  g0[2] = (unsigned int)ga;                                    // global_addr[31:0]
  g0[3] = (unsigned int)((ga >> 32) & 0x01FFFFFFu) | (2u << 30); // addr[56:32]|type=2
  v8u g1;
  g1[0] = 1u << 16;                                            // data_size=1 (2B)
  g1[1] = (tensor_d0 & 0xFFFFu) << 16;                         // tensor_dim0[15:0]
  g1[2] = (tensor_d0 >> 16) | ((tensor_d1 & 0xFFFFu) << 16);   // d0 hi | d1 lo
  g1[3] = (tensor_d1 >> 16) | (tile_d0 << 16);                 // d1 hi | tile_dim0
  g1[4] = tile_d1 & 0xFFFFu;                                   // tile_dim1 (dim2=0)
  g1[5] = stride0_elems;                                       // dim0_stride[31:0]
  g1[6] = 0u;                                                  // stride hi bits
  g1[7] = 0u;
  asm volatile("tensor_load_to_lds %0, %1" :: "s"(g0), "s"(g1) : "memory");
}

// ---- Fused-epilogue GEMM:  C(MxN) = A(MxK,f16) @ W(NxK,f16)^T ---------------
// Block: 256 threads (8 waves, 2M x 4N), tile 32x128, K-step 64.
// W panel (128x64, 16KB) via TDM; A tile (32x64, 4KB) via async global->LDS.
// EPI: 0 = store f16 ; 1 = GELU -> f16 ; 2 = fp32 residual += ; 3 = store fp32
template <int EPI>
__global__ __launch_bounds__(256, 1) void gemm_kernel(
    const _Float16* __restrict__ A, const _Float16* __restrict__ W,
    _Float16* __restrict__ Ch, float* __restrict__ Cf, int M, int N, int K) {
  constexpr int BK = 64, BN = 128, BM = 32;
  __shared__ __align__(32) _Float16 lw[2][BN * BK];   // 32 KB weight panels
  __shared__ __align__(32) _Float16 la[2][BM * BK];   //  8 KB activation tiles

  const int tid = threadIdx.x;
  const int wave = tid >> 5, lane = tid & 31, grp = lane >> 4, ln = lane & 15;
  const int m0 = blockIdx.y * BM + (wave >> 2) * 16;
  const int nblk = blockIdx.x * BN;
  const int n0 = nblk + (wave & 3) * 32;

  // async A-copy role: each of 256 threads moves one 16B chunk per K-step
  const int ar = tid >> 3;                  // 0..31 : row within A tile
  const int ac = (tid & 7) * 8;             // 0..56 : half-chunk within K-step
  const _Float16* aga = A + (size_t)(blockIdx.y * BM + ar) * K + ac;
  const unsigned int aoff = (unsigned int)((ar * BK + ac) * sizeof(_Float16));

  const _Float16* wpanel = W + (size_t)nblk * K;

  const int steps = K / BK;
  auto issue = [&](int s, int buf) {
    async_copy_b128((unsigned int)(uintptr_t)&la[buf][0] + aoff, aga + (size_t)s * BK);
    if (wave == 0)
      tdm_load_tile((unsigned int)(uintptr_t)&lw[buf][0], wpanel + (size_t)s * BK,
                    (unsigned int)K, (unsigned int)N, BK, BN, (unsigned int)K);
  };

  v8f acc0 = {}; v8f acc1 = {};
  issue(0, 0);
  wait_asynccnt0();
  if (wave == 0) __builtin_amdgcn_s_wait_tensorcnt(0);
  __syncthreads();

  const int arow = (wave >> 2) * 16 + ln;
  const int nr0  = (wave & 3) * 32 + ln;
  for (int s = 0; s < steps; ++s) {
    const int cur = s & 1, nxt = cur ^ 1;
    if (s + 1 < steps) issue(s + 1, nxt);        // prefetch next panel via DMA
    const _Float16* laa = &la[cur][0];
    const _Float16* lwa = &lw[cur][0];
#pragma unroll
    for (int kk = 0; kk < 2; ++kk) {             // two 16x16x32 WMMAs per acc
      v16h a  = make_a_frag(laa + arow * BK + kk * 32 + grp * 16);
      v16h b0 = *(const v16h*)(lwa + (size_t)nr0 * BK + kk * 32 + grp * 16);
      v16h b1 = *(const v16h*)(lwa + (size_t)(nr0 + 16) * BK + kk * 32 + grp * 16);
      acc0 = wmma_f16(a, b0, acc0);
      acc1 = wmma_f16(a, b1, acc1);
    }
    if (s + 1 < steps) {
      wait_asynccnt0();                          // my wave's A chunks landed
      if (wave == 0) __builtin_amdgcn_s_wait_tensorcnt(0);  // W panel landed
    }
    __syncthreads();
  }
  (void)M;
#pragma unroll
  for (int i = 0; i < 8; ++i) {
    const size_t m = (size_t)m0 + i + 8 * grp;
    const size_t n = (size_t)n0 + ln;
    float v0 = acc0[i], v1 = acc1[i];
    if constexpr (EPI == 1) {
      v0 = 0.5f * v0 * (1.0f + erff(v0 * 0.70710678f));
      v1 = 0.5f * v1 * (1.0f + erff(v1 * 0.70710678f));
    }
    if constexpr (EPI == 0 || EPI == 1) {
      Ch[m * N + n]      = (_Float16)v0;
      Ch[m * N + n + 16] = (_Float16)v1;
    } else if constexpr (EPI == 2) {
      Cf[m * N + n]      += v0;
      Cf[m * N + n + 16] += v1;
    } else {
      Cf[m * N + n]      = v0;
      Cf[m * N + n + 16] = v1;
    }
  }
}

// ---- Flash attention: one wave per (batch, head, 16-row Q tile) ------------
__global__ __launch_bounds__(32, 1) void attn_kernel(
    const _Float16* __restrict__ qkv, _Float16* __restrict__ out) {
  const int qt = blockIdx.x, h = blockIdx.y, b = blockIdx.z;
  const int lane = threadIdx.x & 31;
  const int grp = lane >> 4, ln = lane & 15;
  const int q0 = qt * 16;

  const _Float16* base  = qkv + (size_t)b * SEQ * QKVD;
  const _Float16* qbase = base + h * HDIM;
  const _Float16* kbase = base + DMODEL + h * HDIM;
  const _Float16* vbase = base + 2 * DMODEL + h * HDIM;

  v16h aq[2];
#pragma unroll
  for (int kk = 0; kk < 2; ++kk)
    aq[kk] = make_a_frag(qbase + (size_t)(q0 + ln) * QKVD + kk * 32 + grp * 16);

  v8f o0 = {}, o1 = {}, o2 = {}, o3 = {};
  float rmax[8], rsum[8];
#pragma unroll
  for (int i = 0; i < 8; ++i) { rmax[i] = -1e30f; rsum[i] = 0.f; }

  __shared__ __align__(32) _Float16 pl[16 * 32];
  const float scale = 0.125f;   // 1/sqrt(64)

  for (int j0 = 0; j0 <= q0 + 15; j0 += 32) {
    v8f s0 = {}, s1 = {};
#pragma unroll
    for (int kk = 0; kk < 2; ++kk) {
      v16h bk0 = *(const v16h*)(kbase + (size_t)(j0 + ln) * QKVD + kk * 32 + grp * 16);
      v16h bk1 = *(const v16h*)(kbase + (size_t)(j0 + 16 + ln) * QKVD + kk * 32 + grp * 16);
      s0 = wmma_f16(aq[kk], bk0, s0);
      s1 = wmma_f16(aq[kk], bk1, s1);
    }
    float p0[8], p1[8];
#pragma unroll
    for (int i = 0; i < 8; ++i) {
      const int m = q0 + i + 8 * grp;
      float a = s0[i] * scale, c = s1[i] * scale;
      if (j0 + ln > m)      a = -1e30f;
      if (j0 + 16 + ln > m) c = -1e30f;
      float mx = fmaxf(a, c);
#pragma unroll
      for (int off = 1; off < 16; off <<= 1) mx = fmaxf(mx, __shfl_xor(mx, off, 16));
      const float nm = fmaxf(rmax[i], mx);
      const float resc = __expf(rmax[i] - nm);
      rmax[i] = nm;
      const float e0 = __expf(a - nm), e1 = __expf(c - nm);
      p0[i] = e0; p1[i] = e1;
      float ps = e0 + e1;
#pragma unroll
      for (int off = 1; off < 16; off <<= 1) ps += __shfl_xor(ps, off, 16);
      rsum[i] = rsum[i] * resc + ps;
      o0[i] *= resc; o1[i] *= resc; o2[i] *= resc; o3[i] *= resc;
    }
    __syncthreads();
#pragma unroll
    for (int i = 0; i < 8; ++i) {
      pl[(i + 8 * grp) * 32 + ln]      = (_Float16)p0[i];
      pl[(i + 8 * grp) * 32 + 16 + ln] = (_Float16)p1[i];
    }
    __syncthreads();
    v16h ap = make_a_frag(&pl[ln * 32 + grp * 16]);
#pragma unroll
    for (int t = 0; t < 4; ++t) {
      v16h bv;
#pragma unroll
      for (int kk = 0; kk < 16; ++kk)
        bv[kk] = vbase[(size_t)(j0 + grp * 16 + kk) * QKVD + t * 16 + ln];
      if (t == 0)      o0 = wmma_f16(ap, bv, o0);
      else if (t == 1) o1 = wmma_f16(ap, bv, o1);
      else if (t == 2) o2 = wmma_f16(ap, bv, o2);
      else             o3 = wmma_f16(ap, bv, o3);
    }
    __syncthreads();
  }
  _Float16* orow = out + (size_t)b * SEQ * DMODEL + h * HDIM;
#pragma unroll
  for (int i = 0; i < 8; ++i) {
    const size_t m = (size_t)q0 + i + 8 * grp;
    const float inv = 1.0f / rsum[i];
    orow[m * DMODEL + 0  + ln] = (_Float16)(o0[i] * inv);
    orow[m * DMODEL + 16 + ln] = (_Float16)(o1[i] * inv);
    orow[m * DMODEL + 32 + ln] = (_Float16)(o2[i] * inv);
    orow[m * DMODEL + 48 + ln] = (_Float16)(o3[i] * inv);
  }
}

// ---- LayerNorm -------------------------------------------------------------
__global__ __launch_bounds__(256, 1) void ln_kernel(
    const float* __restrict__ x, const float* __restrict__ w,
    const float* __restrict__ bias, _Float16* __restrict__ out) {
  const int row = blockIdx.x;
  const float* xr = x + (size_t)row * DMODEL;
  float s = 0.f, s2 = 0.f;
  for (int i = threadIdx.x; i < DMODEL; i += 256) {
    float v = xr[i]; s += v; s2 += v * v;
  }
#pragma unroll
  for (int off = 16; off >= 1; off >>= 1) {
    s += __shfl_xor(s, off, 32); s2 += __shfl_xor(s2, off, 32);
  }
  __shared__ float sh[16];
  __shared__ float stats[2];
  const int wave = threadIdx.x >> 5;
  if ((threadIdx.x & 31) == 0) { sh[wave] = s; sh[8 + wave] = s2; }
  __syncthreads();
  if (threadIdx.x == 0) {
    float ts = 0.f, ts2 = 0.f;
    for (int i = 0; i < 8; ++i) { ts += sh[i]; ts2 += sh[8 + i]; }
    const float mu = ts / DMODEL;
    const float var = ts2 / DMODEL - mu * mu;
    stats[0] = mu; stats[1] = rsqrtf(var + 1e-5f);
  }
  __syncthreads();
  const float mu = stats[0], rstd = stats[1];
  for (int i = threadIdx.x; i < DMODEL; i += 256)
    out[(size_t)row * DMODEL + i] = (_Float16)((xr[i] - mu) * rstd * w[i] + bias[i]);
}

// ---- Embedding -------------------------------------------------------------
__global__ __launch_bounds__(256, 1) void embed_kernel(
    const int* __restrict__ ids, const float* __restrict__ tok,
    const float* __restrict__ pos, float* __restrict__ x) {
  const int row = blockIdx.x;
  const int t = row % SEQ;
  const int id = ids[row];
  for (int i = threadIdx.x; i < DMODEL; i += 256)
    x[(size_t)row * DMODEL + i] =
        tok[(size_t)id * DMODEL + i] + pos[(size_t)t * DMODEL + i];
}

// ---- fp32 -> f16 conversion ------------------------------------------------
__global__ __launch_bounds__(256, 1) void cvt_kernel(
    const float* __restrict__ in, _Float16* __restrict__ outp, size_t n) {
  size_t i = (size_t)blockIdx.x * 256 + threadIdx.x;
  const size_t stride = (size_t)gridDim.x * 256;
  for (; i < n; i += stride) outp[i] = (_Float16)in[i];
}

// ---------------------------------------------------------------------------
extern "C" void kernel_launch(void* const* d_in, const int* in_sizes, int n_in,
                              void* d_out, int out_size, void* d_ws, size_t ws_size,
                              hipStream_t stream) {
  (void)in_sizes; (void)n_in; (void)out_size; (void)ws_size;
  const int*   ids  = (const int*)d_in[0];
  const float* tok  = (const float*)d_in[1];
  const float* pos  = (const float*)d_in[2];
  const float* ln1w = (const float*)d_in[3];
  const float* ln1b = (const float*)d_in[4];
  const float* qkvw = (const float*)d_in[5];
  const float* outw = (const float*)d_in[6];
  const float* ln2w = (const float*)d_in[7];
  const float* ln2b = (const float*)d_in[8];
  const float* upw  = (const float*)d_in[9];
  const float* dnw  = (const float*)d_in[10];
  const float* lnfw = (const float*)d_in[11];
  const float* lnfb = (const float*)d_in[12];
  float* logits = (float*)d_out;

  size_t off = 0;
  auto allocH = [&](size_t n) -> _Float16* {
    _Float16* p = (_Float16*)((char*)d_ws + off);
    off += ((n * sizeof(_Float16) + 255) & ~(size_t)255);
    return p;
  };
  auto allocF = [&](size_t n) -> float* {
    float* p = (float*)((char*)d_ws + off);
    off += ((n * sizeof(float) + 255) & ~(size_t)255);
    return p;
  };
  _Float16* wqkv  = allocH((size_t)LAYERS * QKVD * DMODEL);
  _Float16* wout  = allocH((size_t)LAYERS * DMODEL * DMODEL);
  _Float16* wup   = allocH((size_t)LAYERS * DFF * DMODEL);
  _Float16* wdn   = allocH((size_t)LAYERS * DMODEL * DFF);
  _Float16* wemb  = allocH((size_t)VOCAB * DMODEL);
  float*    xf    = allocF((size_t)NTOK * DMODEL);
  _Float16* h16   = allocH((size_t)NTOK * DMODEL);
  _Float16* qkv16 = allocH((size_t)NTOK * QKVD);
  _Float16* att16 = allocH((size_t)NTOK * DMODEL);
  _Float16* ff16  = allocH((size_t)NTOK * DFF);

  auto cvt = [&](const float* src, _Float16* dst, size_t n) {
    int blocks = (int)((n + 256 * 8 - 1) / (256 * 8));
    cvt_kernel<<<blocks, 256, 0, stream>>>(src, dst, n);
  };
  cvt(qkvw, wqkv, (size_t)LAYERS * QKVD * DMODEL);
  cvt(outw, wout, (size_t)LAYERS * DMODEL * DMODEL);
  cvt(upw,  wup,  (size_t)LAYERS * DFF * DMODEL);
  cvt(dnw,  wdn,  (size_t)LAYERS * DMODEL * DFF);
  cvt(tok,  wemb, (size_t)VOCAB * DMODEL);

  embed_kernel<<<NTOK, 256, 0, stream>>>(ids, tok, pos, xf);

  const dim3 gQKV(QKVD / 128,  NTOK / 32);
  const dim3 gOUT(DMODEL / 128, NTOK / 32);
  const dim3 gUP (DFF / 128,   NTOK / 32);
  const dim3 gLM (VOCAB / 128, NTOK / 32);
  const dim3 gATT(SEQ / 16, HEADS, BATCH);

  for (int l = 0; l < LAYERS; ++l) {
    ln_kernel<<<NTOK, 256, 0, stream>>>(xf, ln1w + l * DMODEL, ln1b + l * DMODEL, h16);
    gemm_kernel<0><<<gQKV, 256, 0, stream>>>(
        h16, wqkv + (size_t)l * QKVD * DMODEL, qkv16, nullptr, NTOK, QKVD, DMODEL);
    attn_kernel<<<gATT, 32, 0, stream>>>(qkv16, att16);
    gemm_kernel<2><<<gOUT, 256, 0, stream>>>(
        att16, wout + (size_t)l * DMODEL * DMODEL, nullptr, xf, NTOK, DMODEL, DMODEL);
    ln_kernel<<<NTOK, 256, 0, stream>>>(xf, ln2w + l * DMODEL, ln2b + l * DMODEL, h16);
    gemm_kernel<1><<<gUP, 256, 0, stream>>>(
        h16, wup + (size_t)l * DFF * DMODEL, ff16, nullptr, NTOK, DFF, DMODEL);
    gemm_kernel<2><<<gOUT, 256, 0, stream>>>(
        ff16, wdn + (size_t)l * DMODEL * DFF, nullptr, xf, NTOK, DMODEL, DFF);
  }
  ln_kernel<<<NTOK, 256, 0, stream>>>(xf, lnfw, lnfb, h16);
  gemm_kernel<3><<<gLM, 256, 0, stream>>>(
      h16, wemb, nullptr, logits, NTOK, VOCAB, DMODEL);
}
